// DPRNNSeparator_78108275245645
// MI455X (gfx1250) — matrix operand order repaired and checked by reading, not compiled, not run on hardware
//
#include <hip/hip_runtime.h>
#include <hip/hip_bf16.h>
#include <math.h>

// ---------------- problem constants ----------------
constexpr int B_    = 2;
constexpr int ENC_  = 64;
constexpr int BOT_  = 64;
constexpr int T_    = 8000;
constexpr int K_    = 100;
constexpr int STEP_ = 50;
constexpr int S_    = 161;            // L = 8100, (L-K)/STEP + 1
constexpr int NB_   = 6;
constexpr int SPK_  = 5;
constexpr int H_    = 64;
constexpr int G4H_  = 4 * H_;         // 256 gate outputs
constexpr long NT_  = 32200;          // N*Tlen, same for intra (322*100) and inter (200*161)
constexpr long PERB_= 16100;          // NT_ / B_
constexpr float EPS_ = 1e-8f;

// ---------------- workspace layout (bytes) ----------------
constexpr size_t OFF_STATS = 0;                                  // 64 floats
constexpr size_t OFF_XT    = 256;                                // [B][T][ENC] f16
constexpr size_t OFF_XBOT  = OFF_XT    + (size_t)B_*T_*ENC_*2;   // [B][T][BOT] f32
constexpr size_t OFF_CHUNK = OFF_XBOT  + (size_t)B_*T_*BOT_*4;   // [B][S][K][C] f32
constexpr size_t OFF_XSEQ  = OFF_CHUNK + (size_t)B_*S_*K_*BOT_*4;// [N][Tlen][C] f16
constexpr size_t OFF_PRE   = OFF_XSEQ  + (size_t)NT_*BOT_*2;     // [2][N][Tlen][256] f16
constexpr size_t OFF_HOUT  = OFF_PRE   + (size_t)2*NT_*G4H_*2;   // [N][Tlen][128] f16
constexpr size_t OFF_Y     = OFF_HOUT  + (size_t)NT_*2*H_*2;     // [M][64] f32
constexpr size_t OFF_YACT  = OFF_Y     + (size_t)NT_*BOT_*4;     // [B][T][64] f16

typedef __attribute__((ext_vector_type(16))) _Float16 v16h;
typedef __attribute__((ext_vector_type(4)))  _Float16 v4h;
typedef __attribute__((ext_vector_type(8)))  float    v8f;
typedef int v4i_ __attribute__((vector_size(16)));   // builtin's pointee type

// CDNA5 async global->LDS copy path (guarded: falls back to direct loads)
#if __has_builtin(__builtin_amdgcn_global_load_async_to_lds_b128) && \
    __has_builtin(__builtin_amdgcn_s_wait_asynccnt)
#define USE_ASYNC_LDS 1
#else
#define USE_ASYNC_LDS 0
#endif

#if USE_ASYNC_LDS
typedef __attribute__((address_space(1))) v4i_* gv4i_p;
typedef __attribute__((address_space(3))) v4i_* lv4i_p;
#endif

__device__ __forceinline__ float sigf(float x) { return 1.0f / (1.0f + __expf(-x)); }

// ============================================================
// Generic WMMA GEMM:  C[M][N] = A[M][K] (f16, row-major) x W[N][K]^T (f32 weights)
// optional bias[N]; write f32 C and/or f16 C; olayout==1 -> speaker-strided store
// ============================================================
#define GEMM_WAVES 4
__global__ void wmma_gemm_wt_kernel(const _Float16* __restrict__ A,
                                    const float*    __restrict__ W,
                                    const float*    __restrict__ bias,
                                    float*          __restrict__ Cf,
                                    _Float16*       __restrict__ Ch,
                                    int M, int Ksz, int Nsz,
                                    int olayout, int Tspk)
{
    const int lane  = threadIdx.x & 31;
    const int wid   = threadIdx.x >> 5;
    const int mtile = blockIdx.x * GEMM_WAVES + wid;
    const int ntile = blockIdx.y;
    const int m0 = mtile * 16;
    if (m0 >= M) return;                          // whole wave exits together
    const int n0 = ntile * 16;

    int am = m0 + (lane & 15); if (am >= M) am = M - 1;     // clamp pad rows
    const int bn   = n0 + (lane & 15);
    const int half = lane >> 4;
    const float* wrow = W + (size_t)bn * Ksz;

    v8f acc = {};
    for (int k0 = 0; k0 < Ksz; k0 += 32) {
        const _Float16* ap = A + (size_t)am * Ksz + k0;
        v16h afr, bfr;
#pragma unroll
        for (int e = 0; e < 16; ++e) {            // A 16x32 fragment (row-major gather)
            int v = e >> 1, w = e & 1;
            int kk = (v < 4 ? half * 8 + v * 2 : 16 + half * 8 + (v - 4) * 2) + w;
            afr[e] = ap[kk];
        }
        const int kb = k0 + half * 8;
#pragma unroll
        for (int e = 0; e < 16; ++e)              // B 32x16 fragment = W^T, f32->f16
            bfr[e] = (_Float16)wrow[kb + ((e & 8) << 1) + (e & 7)];
        acc = __builtin_amdgcn_wmma_f32_16x16x32_f16(false, afr, false, bfr,
                                                     (short)0, acc, false, false);
    }

    const int n  = n0 + (lane & 15);
    const float bv = bias ? bias[n] : 0.0f;
    const int mb = m0 + 8 * half;
#pragma unroll
    for (int j = 0; j < 8; ++j) {
        const int m = mb + j;
        if (m >= M) continue;
        const float vv = acc[j] + bv;
        if (olayout == 0) {
            if (Cf) Cf[(size_t)m * Nsz + n] = vv;
            if (Ch) Ch[(size_t)m * Nsz + n] = (_Float16)vv;
        } else {                                   // mask layout: (b, o, t)
            const int bidx = m / Tspk, t = m % Tspk;
            Cf[((size_t)bidx * Nsz + n) * Tspk + t] = vv;
        }
    }
}

// ============================================================
// BiLSTM scan: one block (8 waves) per 16-sequence tile per direction.
// h kept in LDS (f16); Wh B-fragments resident in VGPRs; WMMA per step.
// pre-activation tile for step t+1 is streamed into a double-buffered LDS
// tile with GLOBAL_LOAD_ASYNC_TO_LDS_B128, overlapped with the WMMA + gate
// math of step t, gated by S_WAIT_ASYNCCNT (in-order async completion).
// pre: [2][N][Tlen][256] f16 (x@Wi^T + b), hout: [N][Tlen][128] f16
// ============================================================
__global__ void lstm_scan_kernel(const _Float16* __restrict__ pre,
                                 const float*    __restrict__ Wh,   // [2][256][64]
                                 _Float16*       __restrict__ hout,
                                 int N, int Tlen)
{
    __shared__ _Float16 hl[16][H_ + 8];
    __shared__ float    gl[16][G4H_ + 4];
#if USE_ASYNC_LDS
    __shared__ _Float16 pbuf[2][16][G4H_];        // 2 x 8KB double buffer
#endif

    const int tid  = threadIdx.x;
    const int lane = tid & 31;
    const int wid  = tid >> 5;
    const int dir  = blockIdx.y;
    const int r0   = blockIdx.x * 16;

    const float* whd = Wh + (size_t)dir * G4H_ * H_;

    // Preload Wh fragments: wave owns gate column tiles {2*wid, 2*wid+1}
    v16h bf[2][2];
#pragma unroll
    for (int cti = 0; cti < 2; ++cti) {
        const int nrow = (wid * 2 + cti) * 16 + (lane & 15);
#pragma unroll
        for (int kt = 0; kt < 2; ++kt) {
            const int kb = kt * 32 + (lane >> 4) * 8;
#pragma unroll
            for (int e = 0; e < 16; ++e)
                bf[cti][kt][e] = (_Float16)whd[(size_t)nrow * H_ + kb + ((e & 8) << 1) + (e & 7)];
        }
    }

    for (int i = tid; i < 16 * H_; i += blockDim.x) hl[i >> 6][i & 63] = (_Float16)0.0f;
    float creg[4] = {0.f, 0.f, 0.f, 0.f};

    // gate-phase ownership: 4 consecutive cells of the 16x64 (row, j) grid
    const int gm = (tid * 4) >> 6;                 // row 0..15
    const int gj = (tid * 4) & 63;                 // col base (multiple of 4)
    int rr = r0 + gm;
    const bool rvalid = (rr < N);
    if (rr >= N) rr = N - 1;
    const _Float16* prer = pre + ((size_t)dir * N + rr) * (size_t)Tlen * G4H_;
    (void)prer;

#if USE_ASYNC_LDS
    // async tile copy: thread -> (row = tid/16, 32B segment = tid%16)
    const int arow = tid >> 4;
    const int achk = (tid & 15) * 32;              // byte offset in 512B row
    int ar = r0 + arow; if (ar >= N) ar = N - 1;
    const char* agsrc = (const char*)(pre + ((size_t)dir * N + ar) * (size_t)Tlen * G4H_);

    // prologue: stream the first step's tile into buffer 0
    {
        const int t0 = dir ? (Tlen - 1) : 0;
        const char* g8 = agsrc + (size_t)t0 * (G4H_ * 2) + achk;
        char* l8 = (char*)&pbuf[0][arow][0] + achk;
        __builtin_amdgcn_global_load_async_to_lds_b128(
            (gv4i_p)g8, (lv4i_p)l8, 0, 0);
        __builtin_amdgcn_global_load_async_to_lds_b128(
            (gv4i_p)(g8 + 16), (lv4i_p)(l8 + 16), 0, 0);
    }
#endif
    __syncthreads();

    for (int it = 0; it < Tlen; ++it) {
        const int t = dir ? (Tlen - 1 - it) : it;

#if USE_ASYNC_LDS
        // stream next step's tile into the other buffer (overlaps WMMA+gates)
        {
            int itn = it + 1; if (itn >= Tlen) itn = Tlen - 1;   // keep count invariant
            const int tn = dir ? (Tlen - 1 - itn) : itn;
            const char* g8 = agsrc + (size_t)tn * (G4H_ * 2) + achk;
            char* l8 = (char*)&pbuf[(it + 1) & 1][arow][0] + achk;
            __builtin_amdgcn_global_load_async_to_lds_b128(
                (gv4i_p)g8, (lv4i_p)l8, 0, 0);
            __builtin_amdgcn_global_load_async_to_lds_b128(
                (gv4i_p)(g8 + 16), (lv4i_p)(l8 + 16), 0, 0);
        }
#else
        if (it + 1 < Tlen) {                       // fallback: cache prefetch
            const int tn = dir ? (t - 1) : (t + 1);
            __builtin_prefetch(prer + (size_t)tn * G4H_ + gj, 0, 1);
        }
#endif

        // A fragments from LDS h (16x64 split into two 16x32 k-tiles)
        const int mrow = lane & 15, half = lane >> 4;
        v16h a0, a1;
#pragma unroll
        for (int e = 0; e < 16; ++e) {
            int v = e >> 1, w = e & 1;
            int kk = (v < 4 ? half * 8 + v * 2 : 16 + half * 8 + (v - 4) * 2) + w;
            a0[e] = hl[mrow][kk];
            a1[e] = hl[mrow][32 + kk];
        }

#pragma unroll
        for (int cti = 0; cti < 2; ++cti) {
            const int ct = wid * 2 + cti;
            const int n  = ct * 16 + (lane & 15);
            const int mb = 8 * (lane >> 4);
            v8f acc = {};
            acc = __builtin_amdgcn_wmma_f32_16x16x32_f16(false, a0, false, bf[cti][0],
                                                         (short)0, acc, false, false);
            acc = __builtin_amdgcn_wmma_f32_16x16x32_f16(false, a1, false, bf[cti][1],
                                                         (short)0, acc, false, false);
#pragma unroll
            for (int j = 0; j < 8; ++j) gl[mb + j][n] = acc[j];
        }

#if USE_ASYNC_LDS
        // leave newest 2 async loads (next tile) in flight; current tile done
        __builtin_amdgcn_s_wait_asynccnt(2);
        __syncthreads();
        const _Float16* pt = &pbuf[it & 1][gm][0];
        const v4h pi = *(const v4h*)(pt + 0 * H_ + gj);
        const v4h pf = *(const v4h*)(pt + 1 * H_ + gj);
        const v4h pg = *(const v4h*)(pt + 2 * H_ + gj);
        const v4h po = *(const v4h*)(pt + 3 * H_ + gj);
#else
        // coalesced vector loads of this thread's 4 pre-activation groups
        const _Float16* pt = prer + (size_t)t * G4H_;
        const v4h pi = *(const v4h*)(pt + 0 * H_ + gj);
        const v4h pf = *(const v4h*)(pt + 1 * H_ + gj);
        const v4h pg = *(const v4h*)(pt + 2 * H_ + gj);
        const v4h po = *(const v4h*)(pt + 3 * H_ + gj);
        __syncthreads();
#endif

        // gates: recurrent term from LDS (float4), pre term added elementwise
        const float4 ri = *(const float4*)&gl[gm][0 * H_ + gj];
        const float4 rf = *(const float4*)&gl[gm][1 * H_ + gj];
        const float4 rg = *(const float4*)&gl[gm][2 * H_ + gj];
        const float4 ro = *(const float4*)&gl[gm][3 * H_ + gj];
        v4h hv;
#pragma unroll
        for (int q = 0; q < 4; ++q) {
            const float iv = ((const float*)&ri)[q] + (float)pi[q];
            const float fv = ((const float*)&rf)[q] + (float)pf[q];
            const float gv = ((const float*)&rg)[q] + (float)pg[q];
            const float ov = ((const float*)&ro)[q] + (float)po[q];
            const float c  = sigf(fv) * creg[q] + sigf(iv) * tanhf(gv);
            creg[q] = c;
            hv[q] = (_Float16)(sigf(ov) * tanhf(c));
        }
        *(v4h*)&hl[gm][gj] = hv;
        if (rvalid)
            *(v4h*)&hout[((size_t)(r0 + gm) * Tlen + t) * (2 * H_) + dir * H_ + gj] = hv;
        __syncthreads();
    }
}

// ============================================================
// small elementwise / reduction kernels
// ============================================================
__global__ void zero_stats_kernel(float* stats) {
    if (threadIdx.x < 64) stats[threadIdx.x] = 0.0f;
}

__global__ void reduce_stats_kernel(const float* __restrict__ x, long per_b,
                                    float* __restrict__ stats)
{
    const int b = blockIdx.y;
    const float* xb = x + (size_t)b * per_b;
    float s = 0.f, s2 = 0.f;
    for (long i = (long)blockIdx.x * blockDim.x + threadIdx.x; i < per_b;
         i += (long)gridDim.x * blockDim.x) {
        const float v = xb[i]; s += v; s2 += v * v;
    }
    __shared__ float ss[256], sq[256];
    ss[threadIdx.x] = s; sq[threadIdx.x] = s2; __syncthreads();
    for (int o = 128; o > 0; o >>= 1) {
        if (threadIdx.x < o) { ss[threadIdx.x] += ss[threadIdx.x + o];
                               sq[threadIdx.x] += sq[threadIdx.x + o]; }
        __syncthreads();
    }
    if (threadIdx.x == 0) {
        atomicAdd(&stats[b * 2 + 0], ss[0]);
        atomicAdd(&stats[b * 2 + 1], sq[0]);
    }
}

__global__ void front_norm_kernel(const float* __restrict__ enc,
                                  const float* __restrict__ g,
                                  const float* __restrict__ be,
                                  const float* __restrict__ stats,
                                  _Float16*    __restrict__ xt)
{
    const long i = (long)blockIdx.x * blockDim.x + threadIdx.x;
    if (i >= (long)B_ * ENC_ * T_) return;
    const int t = (int)(i % T_);
    const int n = (int)((i / T_) % ENC_);
    const int b = (int)(i / ((long)ENC_ * T_));
    const float cnt = (float)ENC_ * (float)T_;
    const float mu  = stats[b * 2] / cnt;
    const float var = stats[b * 2 + 1] / cnt - mu * mu;
    const float rs  = rsqrtf(var + EPS_);
    const float v   = (enc[i] - mu) * rs * g[n] + be[n];
    xt[((size_t)b * T_ + t) * ENC_ + n] = (_Float16)v;
}

__global__ void chunk_build_kernel(const float* __restrict__ xbot,
                                   float* __restrict__ chunks)
{
    const long i = (long)blockIdx.x * blockDim.x + threadIdx.x;
    if (i >= (long)B_ * S_ * K_ * BOT_) return;
    const int c = (int)(i & 63);
    long r = i >> 6;
    const int k = (int)(r % K_); r /= K_;
    const int s = (int)(r % S_);
    const int b = (int)(r / S_);
    const int t = s * STEP_ + k - STEP_;
    chunks[i] = (t >= 0 && t < T_) ? xbot[((size_t)b * T_ + t) * BOT_ + c] : 0.0f;
}

__global__ void gather_seq_kernel(const float* __restrict__ chunks,
                                  _Float16* __restrict__ xseq, int mode)
{
    const long i = (long)blockIdx.x * blockDim.x + threadIdx.x;
    if (i >= (long)B_ * S_ * K_ * BOT_) return;
    const int c = (int)(i & 63);
    long r = i >> 6;
    const int k = (int)(r % K_); r /= K_;
    const int s = (int)(r % S_);
    const int b = (int)(r / S_);
    const _Float16 v = (_Float16)chunks[i];
    if (mode == 0) xseq[i] = v;                                   // [(b*S+s)][k][c]
    else xseq[(((size_t)b * K_ + k) * S_ + s) * BOT_ + c] = v;    // [(b*K+k)][s][c]
}

__global__ void apply_gn_res_kernel(const float* __restrict__ y,
                                    const float* __restrict__ stats,
                                    const float* __restrict__ g,
                                    const float* __restrict__ be,
                                    float* __restrict__ chunks, int mode)
{
    const long i = (long)blockIdx.x * blockDim.x + threadIdx.x;
    if (i >= (long)B_ * S_ * K_ * BOT_) return;
    const int c = (int)(i & 63);
    long r = i >> 6;
    const int k = (int)(r % K_); r /= K_;
    const int s = (int)(r % S_);
    const int b = (int)(r / S_);
    const long m = (mode == 0) ? ((long)(b * S_ + s) * K_ + k)
                               : ((long)(b * K_ + k) * S_ + s);
    const float cnt = (float)PERB_ * (float)BOT_;
    const float mu  = stats[b * 2] / cnt;
    const float var = stats[b * 2 + 1] / cnt - mu * mu;
    const float rs  = rsqrtf(var + EPS_);
    chunks[i] += (y[m * BOT_ + c] - mu) * rs * g[c] + be[c];
}

__global__ void ola_prelu_kernel(const float* __restrict__ chunks,
                                 const float* __restrict__ prelu_a,
                                 _Float16* __restrict__ yact)
{
    const long i = (long)blockIdx.x * blockDim.x + threadIdx.x;
    if (i >= (long)B_ * T_ * BOT_) return;
    const int c = (int)(i & 63);
    long r = i >> 6;
    const int t = (int)(r % T_);
    const int b = (int)(r / T_);
    const int l = t + STEP_;
    int smin = (l - K_) / STEP_; if (smin < 0) smin = 0;
    int smax = l / STEP_;        if (smax > S_ - 1) smax = S_ - 1;
    float sum = 0.f, cnt = 0.f;
    for (int s = smin; s <= smax; ++s) {
        const int k = l - s * STEP_;
        if (k >= 0 && k < K_) {
            sum += chunks[(((size_t)b * S_ + s) * K_ + k) * BOT_ + c];
            cnt += 1.f;
        }
    }
    float v = sum / fmaxf(cnt, 1.f);
    const float a = prelu_a[0];
    v = (v >= 0.f) ? v : a * v;
    yact[((size_t)b * T_ + t) * BOT_ + c] = (_Float16)v;
}

__global__ void softmax_spk_kernel(float* __restrict__ out)
{
    const long i = (long)blockIdx.x * blockDim.x + threadIdx.x;
    if (i >= (long)B_ * ENC_ * T_) return;
    const int t = (int)(i % T_);
    const int n = (int)((i / T_) % ENC_);
    const int b = (int)(i / ((long)ENC_ * T_));
    const size_t base   = (((size_t)b * SPK_) * ENC_ + n) * T_ + t;
    const size_t stride = (size_t)ENC_ * T_;
    float mx = -1e30f;
    float e[SPK_];
#pragma unroll
    for (int s = 0; s < SPK_; ++s) { const float v = out[base + s * stride]; mx = fmaxf(mx, v); }
    float sum = 0.f;
#pragma unroll
    for (int s = 0; s < SPK_; ++s) { e[s] = __expf(out[base + s * stride] - mx); sum += e[s]; }
    const float inv = 1.0f / sum;
#pragma unroll
    for (int s = 0; s < SPK_; ++s) out[base + s * stride] = e[s] * inv;
}

// ============================================================
// host launcher
// ============================================================
static inline void launch_gemm(const _Float16* A, const float* W, const float* bias,
                               float* Cf, _Float16* Ch, int M, int Ksz, int Nsz,
                               int olayout, int Tspk, hipStream_t stream)
{
    const int mtiles = (M + 15) / 16;
    dim3 grid((mtiles + GEMM_WAVES - 1) / GEMM_WAVES, Nsz / 16);
    wmma_gemm_wt_kernel<<<grid, GEMM_WAVES * 32, 0, stream>>>(
        A, W, bias, Cf, Ch, M, Ksz, Nsz, olayout, Tspk);
}

extern "C" void kernel_launch(void* const* d_in, const int* in_sizes, int n_in,
                              void* d_out, int out_size, void* d_ws, size_t ws_size,
                              hipStream_t stream)
{
    const float* encoded  = (const float*)d_in[0];
    const float* gn_gamma = (const float*)d_in[1];
    const float* gn_beta  = (const float*)d_in[2];
    const float* bw       = (const float*)d_in[3];
    const float* bb       = (const float*)d_in[4];
    const float* blk_Wi   = (const float*)d_in[5];
    const float* blk_Wh   = (const float*)d_in[6];
    const float* blk_b    = (const float*)d_in[7];
    const float* blk_pw   = (const float*)d_in[8];
    const float* blk_pb   = (const float*)d_in[9];
    const float* blk_gg   = (const float*)d_in[10];
    const float* blk_gb   = (const float*)d_in[11];
    const float* prelu_a  = (const float*)d_in[12];
    const float* mask_w   = (const float*)d_in[13];
    const float* mask_b   = (const float*)d_in[14];
    float* out = (float*)d_out;

    char* ws = (char*)d_ws;
    float*    stats  = (float*)   (ws + OFF_STATS);
    _Float16* xt     = (_Float16*)(ws + OFF_XT);
    float*    xbot   = (float*)   (ws + OFF_XBOT);
    float*    chunks = (float*)   (ws + OFF_CHUNK);
    _Float16* xseq   = (_Float16*)(ws + OFF_XSEQ);
    _Float16* pre    = (_Float16*)(ws + OFF_PRE);
    _Float16* hout   = (_Float16*)(ws + OFF_HOUT);
    float*    ybuf   = (float*)   (ws + OFF_Y);
    _Float16* yact   = (_Float16*)(ws + OFF_YACT);

    auto blocks = [](long n) { return (int)((n + 255) / 256); };

    // ---- front: group norm over encoded, then bottleneck GEMM ----
    zero_stats_kernel<<<1, 64, 0, stream>>>(stats);
    reduce_stats_kernel<<<dim3(256, B_), 256, 0, stream>>>(encoded, (long)ENC_ * T_, stats);
    front_norm_kernel<<<blocks((long)B_ * ENC_ * T_), 256, 0, stream>>>(
        encoded, gn_gamma, gn_beta, stats, xt);
    launch_gemm(xt, bw, bb, xbot, nullptr, B_ * T_, ENC_, BOT_, 0, 0, stream);
    chunk_build_kernel<<<blocks((long)B_ * S_ * K_ * BOT_), 256, 0, stream>>>(xbot, chunks);

    // ---- 6 dual-path blocks ----
    for (int i = 0; i < NB_; ++i) {
        for (int j = 0; j < 2; ++j) {                  // 0 = intra(K), 1 = inter(S)
            const int ij   = i * 2 + j;
            const int Tlen = j ? S_ : K_;
            const int Nseq = j ? (B_ * K_) : (B_ * S_);

            gather_seq_kernel<<<blocks((long)B_ * S_ * K_ * BOT_), 256, 0, stream>>>(
                chunks, xseq, j);

            for (int d = 0; d < 2; ++d)                 // input pre-GEMM per direction
                launch_gemm(xseq,
                            blk_Wi + (size_t)(ij * 2 + d) * G4H_ * BOT_,
                            blk_b  + (size_t)(ij * 2 + d) * G4H_,
                            nullptr, pre + (size_t)d * NT_ * G4H_,
                            (int)NT_, BOT_, G4H_, 0, 0, stream);

            lstm_scan_kernel<<<dim3((Nseq + 15) / 16, 2), 256, 0, stream>>>(
                pre, blk_Wh + (size_t)ij * 2 * G4H_ * H_, hout, Nseq, Tlen);

            launch_gemm(hout,
                        blk_pw + (size_t)ij * BOT_ * 2 * H_,
                        blk_pb + (size_t)ij * BOT_,
                        ybuf, nullptr, (int)NT_, 2 * H_, BOT_, 0, 0, stream);

            zero_stats_kernel<<<1, 64, 0, stream>>>(stats);
            reduce_stats_kernel<<<dim3(256, B_), 256, 0, stream>>>(
                ybuf, PERB_ * BOT_, stats);
            apply_gn_res_kernel<<<blocks((long)B_ * S_ * K_ * BOT_), 256, 0, stream>>>(
                ybuf, stats, blk_gg + (size_t)ij * BOT_, blk_gb + (size_t)ij * BOT_,
                chunks, j);
        }
    }

    // ---- overlap-add + PReLU + mask GEMM + softmax ----
    ola_prelu_kernel<<<blocks((long)B_ * T_ * BOT_), 256, 0, stream>>>(
        chunks, prelu_a, yact);
    launch_gemm(yact, mask_w, mask_b, out, nullptr, B_ * T_, BOT_, ENC_ * SPK_,
                1, T_, stream);
    softmax_spk_kernel<<<blocks((long)B_ * ENC_ * T_), 256, 0, stream>>>(out);
}
